// MultiHeadGraphAttention_20151986553268
// MI455X (gfx1250) — compile-verified, matrix operand order checked by reading, not run on hardware
//
#include <hip/hip_runtime.h>
#include <hip/hip_bf16.h>
#include <math.h>

#define N_NODES 50000
#define N_EDGES 400000
#define IN_DIM  256
#define HID     64
#define HEADS   8
#define OUT_DIM 256
#define HTOT    (HEADS * HID)   // 512
#define NEG_SLOPE 0.2f

typedef __bf16 bf16_t;
typedef bf16_t v16bf __attribute__((ext_vector_type(16)));
typedef bf16_t v8bf  __attribute__((ext_vector_type(8)));
typedef float  v8f   __attribute__((ext_vector_type(8)));
typedef unsigned u32x8 __attribute__((ext_vector_type(8)));

// ---- helpers ---------------------------------------------------------------

// Pack two f32 -> one u32 holding two RNE-rounded bf16 (lo in [15:0]).
__device__ __forceinline__ unsigned bfpair(float lo, float hi) {
    unsigned ul = __float_as_uint(lo);
    unsigned uh = __float_as_uint(hi);
    ul += 0x7FFFu + ((ul >> 16) & 1u);
    uh += 0x7FFFu + ((uh >> 16) & 1u);
    return (ul >> 16) | (uh & 0xFFFF0000u);
}

// Order-preserving float<->uint map for integer atomicMax segment-max.
__device__ __forceinline__ unsigned f2ord(float f) {
    unsigned u = __float_as_uint(f);
    return (u >> 31) ? ~u : (u | 0x80000000u);
}
__device__ __forceinline__ float ord2f(unsigned o) {
    return __uint_as_float((o >> 31) ? (o & 0x7FFFFFFFu) : ~o);
}
#define ORD_NEG_INF 0x007FFFFFu   // == f2ord(-inf)

// ---- kernel 0: init accumulators ------------------------------------------

__global__ void gat_init(float* __restrict__ acc, unsigned* __restrict__ m_ord,
                         float* __restrict__ denom) {
    int idx = blockIdx.x * blockDim.x + threadIdx.x;
    if (idx < N_NODES * HTOT) acc[idx] = 0.0f;
    if (idx < N_NODES * HEADS) { m_ord[idx] = ORD_NEG_INF; denom[idx] = 0.0f; }
}

// ---- prep: x (f32 row-major) -> bf16 row-major ----------------------------

__global__ void gat_prep_x(const float2* __restrict__ x2, unsigned* __restrict__ xbf,
                           int total2) {
    int idx = blockIdx.x * blockDim.x + threadIdx.x;
    if (idx >= total2) return;
    float2 v = x2[idx];
    xbf[idx] = bfpair(v.x, v.y);
}

// ---- prep: pack a K x N f32 matrix into per-lane WMMA B fragments ---------
// Output layout: [mat][nt][kk][lane(32)][i(16)] bf16, i.e. each lane's 16
// B-fragment values are 32 contiguous bytes; consecutive lanes contiguous.
// B element for (lane,i): N = nt*16 + (lane&15);
//   K = kk*32 + (i<8 ? 8*(lane>>4)+i : 8+8*(lane>>4)+i)   (per ISA 7.12.2)

__global__ void gat_pack_b(const float* __restrict__ src, unsigned* __restrict__ dst,
                           int ksteps, int ntiles, int ld, int matstride, int total2) {
    int p2 = blockIdx.x * blockDim.x + threadIdx.x;
    if (p2 >= total2) return;
    int p    = p2 << 1;            // even element index; pair (i, i+1) same run
    int i    = p & 15;
    int lane = (p >> 4) & 31;
    int f    = p >> 9;             // fragment index = ((mat*ntiles + nt)*ksteps + kk)
    int kk   = f % ksteps; f /= ksteps;
    int nt   = f % ntiles; f /= ntiles;
    int mat  = f;
    int g = lane >> 4, l = lane & 15;
    int k = kk * 32 + ((i < 8) ? (8 * g + i) : (8 + 8 * g + i));
    int n = nt * 16 + l;
    const float* __restrict__ m = src + (size_t)mat * matstride;
    float v0 = m[(size_t)k * ld + n];
    float v1 = m[(size_t)(k + 1) * ld + n];
    dst[p2] = bfpair(v0, v1);
}

// ---- kernel 1: h = x @ W  (per head), bf16 WMMA ---------------------------
// 8 waves/block in a 2(M)x4(N) grid -> 32x64 tile; grid=(ceil(N/32), HEADS).
// Inner loop: 2x b128 A loads + 2x b128 B loads + 1 WMMA, fully unrolled.

__global__ __launch_bounds__(256) void gat_gemm1(const unsigned short* __restrict__ xbf,
                                                 const unsigned short* __restrict__ wpack,
                                                 float* __restrict__ hbuf) {
    const int tid  = threadIdx.x;
    const int lane = tid & 31;
    const int wave = tid >> 5;
    const int g    = lane >> 4;
    const int l    = lane & 15;
    const int head = blockIdx.y;
    const int rm   = blockIdx.x * 32 + (wave >> 2) * 16;
    const int nt   = wave & 3;

    int row = rm + l;
    if (row >= N_NODES) row = 0;                 // clamp; store is guarded
    const v8bf* __restrict__ xr = (const v8bf*)(xbf + (size_t)row * IN_DIM);
    const v16bf* __restrict__ bp = (const v16bf*)wpack;
    const int fragbase = (head * 4 + nt) * 8;    // ksteps = 8

    v8f acc = {};
#pragma unroll
    for (int kk = 0; kk < 8; ++kk) {
        const int c1 = kk * 4 + g;       // 8-elem chunk at K = kk*32 + 8g
        const int c2 = kk * 4 + 2 + g;   // 8-elem chunk at K = kk*32 + 16 + 8g
        v8bf lo = xr[c1];
        v8bf hi = xr[c2];
        v16bf va = __builtin_shufflevector(lo, hi, 0,1,2,3,4,5,6,7,8,9,10,11,12,13,14,15);
        v16bf vb = bp[(size_t)(fragbase + kk) * 32 + lane];
        acc = __builtin_amdgcn_wmma_f32_16x16x32_bf16(
            false, va, false, vb, (short)0, acc, false, false);
    }
    // C/D layout: VGPR v, lane -> (M = v + 8*g, N = lane&15)
#pragma unroll
    for (int v = 0; v < 8; ++v) {
        int r = rm + v + 8 * g;
        if (r < N_NODES)
            hbuf[(size_t)r * HTOT + head * HID + nt * 16 + l] = acc[v];
    }
}

// ---- kernel 2: per-node attention scores ----------------------------------

__global__ void gat_scores(const float* __restrict__ hbuf, const float* __restrict__ a,
                           float* __restrict__ s_src, float* __restrict__ s_tgt) {
    int idx = blockIdx.x * blockDim.x + threadIdx.x;   // n*HEADS + head
    if (idx >= N_NODES * HEADS) return;
    int n = idx >> 3, hd = idx & 7;
    const float4* __restrict__ hr = (const float4*)(hbuf + (size_t)n * HTOT + hd * HID);
    const float4* __restrict__ as = (const float4*)(a + hd * 2 * HID);
    const float4* __restrict__ at = (const float4*)(a + hd * 2 * HID + HID);
    float ss = 0.0f, st = 0.0f;
#pragma unroll
    for (int o = 0; o < HID / 4; ++o) {
        float4 hv = hr[o], av = as[o], tv = at[o];
        ss += hv.x * av.x + hv.y * av.y + hv.z * av.z + hv.w * av.w;
        st += hv.x * tv.x + hv.y * tv.y + hv.z * tv.z + hv.w * tv.w;
    }
    s_src[idx] = ss;
    s_tgt[idx] = st;
}

// ---- kernel 3: edge logits (leaky relu) + segment max ---------------------

__global__ void gat_edge_logits(const int* __restrict__ ei,
                                const float* __restrict__ s_src,
                                const float* __restrict__ s_tgt,
                                float* __restrict__ e_buf,
                                unsigned* __restrict__ m_ord) {
    int idx = blockIdx.x * blockDim.x + threadIdx.x;   // e*HEADS + head
    if (idx >= N_EDGES * HEADS) return;
    int e = idx >> 3, hd = idx & 7;
    int src = ei[e];
    int tgt = ei[N_EDGES + e];
    float v = s_src[src * HEADS + hd] + s_tgt[tgt * HEADS + hd];
    v = (v >= 0.0f) ? v : NEG_SLOPE * v;
    e_buf[idx] = v;
    atomicMax(&m_ord[tgt * HEADS + hd], f2ord(v));
}

// ---- kernel 4: exp(e - max) + segment sum ---------------------------------

__global__ void gat_edge_exp(const int* __restrict__ ei,
                             float* __restrict__ e_buf,
                             const unsigned* __restrict__ m_ord,
                             float* __restrict__ denom) {
    int idx = blockIdx.x * blockDim.x + threadIdx.x;
    if (idx >= N_EDGES * HEADS) return;
    int e = idx >> 3, hd = idx & 7;
    int tgt = ei[N_EDGES + e];
    float m  = ord2f(m_ord[tgt * HEADS + hd]);
    float ex = expf(e_buf[idx] - m);
    e_buf[idx] = ex;
    atomicAdd(&denom[tgt * HEADS + hd], ex);
}

// ---- kernel 5: weighted message scatter (bandwidth-bound part) ------------
// One thread per (edge, channel): 512 consecutive lanes share an edge ->
// coalesced h[src] reads + coalesced f32 atomics resolving in L2 (the 102MB
// accumulator fits the 192MB L2).

__global__ __launch_bounds__(256) void gat_scatter(const int* __restrict__ ei,
                                                   const float* __restrict__ hbuf,
                                                   const float* __restrict__ e_buf,
                                                   const float* __restrict__ denom,
                                                   float* __restrict__ acc) {
    long long gid = (long long)blockIdx.x * blockDim.x + threadIdx.x;
    if (gid >= (long long)N_EDGES * HTOT) return;
    int e  = (int)(gid >> 9);
    int c  = (int)(gid & 511);
    int hd = c >> 6;
    int src = ei[e];
    int tgt = ei[N_EDGES + e];
    float ex    = e_buf[e * HEADS + hd];
    float den   = denom[tgt * HEADS + hd];
    float alpha = ex / (den + 1e-16f);
    atomicAdd(&acc[(size_t)tgt * HTOT + c], hbuf[(size_t)src * HTOT + c] * alpha);
}

// ---- kernel 6: out = elu(acc @ Wo + bo), bf16 WMMA ------------------------
// A tile converted f32->bf16 in-register via packed-pair rounding.

__global__ __launch_bounds__(256) void gat_gemm2(const float* __restrict__ acc_in,
                                                 const unsigned short* __restrict__ wopack,
                                                 const float* __restrict__ bo,
                                                 float* __restrict__ out) {
    const int tid  = threadIdx.x;
    const int lane = tid & 31;
    const int wave = tid >> 5;
    const int g    = lane >> 4;
    const int l    = lane & 15;
    const int rm   = blockIdx.x * 32 + (wave >> 2) * 16;
    const int nt   = blockIdx.y * 4 + (wave & 3);    // 16-col tile index in [0,16)
    const int cn   = nt * 16;

    int row = rm + l;
    if (row >= N_NODES) row = 0;
    const float4* __restrict__ ar = (const float4*)(acc_in + (size_t)row * HTOT);
    const v16bf* __restrict__ bp = (const v16bf*)wopack;

    v8f acc = {};
#pragma unroll
    for (int kk = 0; kk < 16; ++kk) {
        const int q = kk * 8 + 2 * g;   // float4 chunk at K = kk*32 + 8g
        float4 a0 = ar[q];
        float4 a1 = ar[q + 1];
        float4 a2 = ar[q + 4];          // K = kk*32 + 16 + 8g
        float4 a3 = ar[q + 5];
        u32x8 pa = { bfpair(a0.x, a0.y), bfpair(a0.z, a0.w),
                     bfpair(a1.x, a1.y), bfpair(a1.z, a1.w),
                     bfpair(a2.x, a2.y), bfpair(a2.z, a2.w),
                     bfpair(a3.x, a3.y), bfpair(a3.z, a3.w) };
        v16bf va = __builtin_bit_cast(v16bf, pa);
        v16bf vb = bp[(size_t)(nt * 16 + kk) * 32 + lane];
        acc = __builtin_amdgcn_wmma_f32_16x16x32_bf16(
            false, va, false, vb, (short)0, acc, false, false);
    }
    const float bias = bo[cn + l];
#pragma unroll
    for (int v = 0; v < 8; ++v) {
        int r = rm + v + 8 * g;
        if (r < N_NODES) {
            float val = acc[v] + bias;
            val = (val > 0.0f) ? val : (expf(val) - 1.0f);   // ELU
            out[(size_t)r * OUT_DIM + cn + l] = val;
        }
    }
}

// ---- launcher --------------------------------------------------------------

static inline size_t align256(size_t v) { return (v + 255) & ~(size_t)255; }

extern "C" void kernel_launch(void* const* d_in, const int* in_sizes, int n_in,
                              void* d_out, int out_size, void* d_ws, size_t ws_size,
                              hipStream_t stream) {
    const float* x  = (const float*)d_in[0];
    const int*   ei = (const int*)d_in[1];     // edge_index, [2, N_EDGES]
    const float* W  = (const float*)d_in[2];   // [HEADS, IN_DIM, HID]
    const float* a  = (const float*)d_in[3];   // [HEADS, 2*HID]
    const float* Wo = (const float*)d_in[4];   // [HTOT, OUT_DIM]
    const float* bo = (const float*)d_in[5];   // [OUT_DIM]
    float* out = (float*)d_out;

    // scratch layout (256B-aligned carves)
    char* ws = (char*)d_ws;
    size_t off = 0;
    float* hbuf = (float*)(ws + off); off = align256(off + (size_t)N_NODES * HTOT * 4);
    float* accb = (float*)(ws + off); off = align256(off + (size_t)N_NODES * HTOT * 4);
    float* s_src = (float*)(ws + off); off = align256(off + (size_t)N_NODES * HEADS * 4);
    float* s_tgt = (float*)(ws + off); off = align256(off + (size_t)N_NODES * HEADS * 4);
    unsigned* m_ord = (unsigned*)(ws + off); off = align256(off + (size_t)N_NODES * HEADS * 4);
    float* denom = (float*)(ws + off); off = align256(off + (size_t)N_NODES * HEADS * 4);
    float* e_buf = (float*)(ws + off); off = align256(off + (size_t)N_EDGES * HEADS * 4);
    unsigned short* xbf    = (unsigned short*)(ws + off); off = align256(off + (size_t)N_NODES * IN_DIM * 2);
    unsigned short* wpack  = (unsigned short*)(ws + off); off = align256(off + (size_t)HEADS * IN_DIM * HID * 2);
    unsigned short* wopack = (unsigned short*)(ws + off); off = align256(off + (size_t)HTOT * OUT_DIM * 2);
    (void)ws_size; (void)in_sizes; (void)n_in; (void)out_size;

    const int T = 256;

    gat_init<<<(N_NODES * HTOT + T - 1) / T, T, 0, stream>>>(accb, m_ord, denom);

    {
        int total2 = N_NODES * IN_DIM / 2;
        gat_prep_x<<<(total2 + T - 1) / T, T, 0, stream>>>((const float2*)x,
                                                           (unsigned*)xbf, total2);
    }
    {   // W: 8 mats of 256x64 (ld=64): ksteps=8, ntiles=4
        int total2 = HEADS * IN_DIM * HID / 2;
        gat_pack_b<<<(total2 + T - 1) / T, T, 0, stream>>>(W, (unsigned*)wpack,
                                                           8, 4, HID, IN_DIM * HID, total2);
    }
    {   // Wo: 1 mat of 512x256 (ld=256): ksteps=16, ntiles=16
        int total2 = HTOT * OUT_DIM / 2;
        gat_pack_b<<<(total2 + T - 1) / T, T, 0, stream>>>(Wo, (unsigned*)wopack,
                                                           16, 16, OUT_DIM, 0, total2);
    }

    gat_gemm1<<<dim3((N_NODES + 31) / 32, HEADS), T, 0, stream>>>(xbf, wpack, hbuf);

    gat_scores<<<(N_NODES * HEADS + T - 1) / T, T, 0, stream>>>(hbuf, a, s_src, s_tgt);

    gat_edge_logits<<<(N_EDGES * HEADS + T - 1) / T, T, 0, stream>>>(ei, s_src, s_tgt,
                                                                     e_buf, m_ord);

    gat_edge_exp<<<(N_EDGES * HEADS + T - 1) / T, T, 0, stream>>>(ei, e_buf, m_ord, denom);

    {
        long long total = (long long)N_EDGES * HTOT;
        int blocks = (int)((total + T - 1) / T);
        gat_scatter<<<blocks, T, 0, stream>>>(ei, hbuf, e_buf, denom, accb);
    }

    gat_gemm2<<<dim3((N_NODES + 31) / 32, OUT_DIM / 64), T, 0, stream>>>(accb, wopack, bo, out);
}
